// SRLEmbeddings_64003602645378
// MI455X (gfx1250) — matrix operand order repaired and checked by reading, not compiled, not run on hardware
//
#include <hip/hip_runtime.h>

typedef __attribute__((ext_vector_type(16))) __bf16          v16bf;
typedef __attribute__((ext_vector_type(8)))  float           v8f;
typedef __attribute__((ext_vector_type(16))) unsigned short  v16us;
typedef __attribute__((ext_vector_type(8)))  unsigned int    v8ui;

#define NBS   256      // B*S workgroups
#define LL    128      // tokens per sentence (K dim)
#define DD    768      // embedding dim (N dim)
#define AA    10
#define TT    8
#define NROW  240      // 3 arg types * A * T candidate rows
#define DC    128      // D-chunk per iteration -> 8 N-tiles = 1 per wave
#define NCH   6        // 6 * 128 = 768
#define RST   134      // u16 row stride (67 dwords): bank-conflict-free for
                       // both transposed staging stores and B-fragment loads
#define OUT0_SZ (NBS*DD)        // 196608  sentence-mean output
#define ARG_SZ  (NBS*AA*DD)     // 1966080 per arg-type output

__device__ __forceinline__ unsigned short f2bf(float f) {
  // round-to-nearest-even f32 -> bf16
  unsigned int u = __builtin_bit_cast(unsigned int, f);
  u += 0x7FFFu + ((u >> 16) & 1u);
  return (unsigned short)(u >> 16);
}

__global__ void __launch_bounds__(256)
srl_fused(const int* __restrict__ sids_g, const float* __restrict__ mask_g,
          const float* __restrict__ emb_g, const int* __restrict__ pred_g,
          const int* __restrict__ a0_g, const int* __restrict__ a1_g,
          float* __restrict__ out)
{
  __shared__ int   sids[LL];
  __shared__ float maskv[LL];
  __shared__ int   aids[NROW];
  __shared__ int   counts[NROW];
  __shared__ int   selT[32];
  __shared__ float rowScale[32];
  __shared__ int   outOfs[32];
  // A matrix stored directly in WMMA 16-bit A-fragment layout:
  // [mt][kk][lane][e] -> one contiguous 32B read per fragment per lane
  __shared__ __align__(32) unsigned short Aswz[2 * 4 * 32 * 16];
  __shared__ __align__(16) unsigned short embT[DC * RST];   // [d][k] bf16

  const int bs   = blockIdx.x;       // (b,s) pair
  const int t    = threadIdx.x;
  const int lane = t & 31;           // wave32
  const int w    = t >> 5;           // 8 waves, wave w owns N-tile w
  const int g    = (lane >> 4) & 1;  // lane half-group

  // ---- phase 0: ids + mask to LDS --------------------------------------
  if (t < LL) { sids[t] = sids_g[bs * LL + t]; maskv[t] = mask_g[bs * LL + t]; }
  if (t < NROW) {
    int at = t / 80, r = t - at * 80;
    const int* p = (at == 0) ? pred_g : (at == 1) ? a0_g : a1_g;
    aids[t] = p[bs * 80 + r];
  }
  __syncthreads();

  // ---- phase 1: match counts per candidate (argtype,a,t) row ------------
  if (t < NROW) {
    int id = aids[t], c = 0;
    if (id != 0)
      for (int l = 0; l < LL; ++l) c += (sids[l] == id) ? 1 : 0;
    counts[t] = c;
  }
  __syncthreads();

  // ---- phase 2: pick LAST t with match; row scales + output offsets -----
  if (t < 30) {
    int at = t / 10, a = t - at * 10;
    int base = at * 80 + a * 8;
    int sel = -1;
    for (int tt = TT - 1; tt >= 0; --tt)
      if (counts[base + tt] > 0) { sel = tt; break; }
    selT[t] = sel;
    int c = (sel >= 0) ? counts[base + sel] : 1;
    rowScale[t] = 1.0f / (float)((c > 1) ? c : 1);
    outOfs[t]   = OUT0_SZ + at * ARG_SZ + (bs * AA + a) * DD;
  } else if (t == 30) {
    float s = 0.f;
    for (int l = 0; l < LL; ++l) s += maskv[l];
    rowScale[30] = 1.0f / fmaxf(s, 1.0f);
    outOfs[30]   = bs * DD;          // sentence mean
    selT[30] = -1;
  } else if (t == 31) {
    rowScale[31] = 0.f; outOfs[31] = 0; selT[31] = -1;   // pad row (never stored)
  }
  __syncthreads();

  // ---- phase 3: build A directly in swizzled fragment layout ------------
  for (int i = t; i < 32 * LL; i += 256) {
    int row = i >> 7, l = i & 127;
    unsigned short v = 0;
    if (row < 30) {
      int at = row / 10, a = row - at * 10;
      int sel = selT[row];
      if (sel >= 0) {
        int id = aids[at * 80 + a * 8 + sel];
        v = (id != 0 && sids[l] == id) ? (unsigned short)0x3F80 : (unsigned short)0;
      }
    } else if (row == 30) {
      v = f2bf(maskv[l]);            // mask-weighted mean row
    }
    // inverse of the 16-bit A 16x32 VGPR layout
    int mt = row >> 4, m = row & 15;
    int kk = l >> 5, r = l & 31;
    int gg, e;
    if (r < 16) { gg = r >> 3; e = r - gg * 8; }
    else        { int rr = r - 16; gg = rr >> 3; e = 8 + (rr - gg * 8); }
    Aswz[(((mt * 4 + kk) * 32) + (gg * 16 + m)) * 16 + e] = v;
  }
  __syncthreads();

  // ---- preload A fragments: one contiguous 32B LDS read each ------------
  v16bf afrag[2][4];
  {
    const v16us* ap = (const v16us*)Aswz;
#pragma unroll
    for (int mt = 0; mt < 2; ++mt)
#pragma unroll
      for (int kk = 0; kk < 4; ++kk)
        afrag[mt][kk] = __builtin_bit_cast(v16bf, ap[(mt * 4 + kk) * 32 + lane]);
  }

  // ---- hoist per-lane C-row scales and output offsets into registers ----
  const int mofs = (lane >= 16) ? 8 : 0;
  float rs0[8], rs1[8]; int of0[8], of1[8];
#pragma unroll
  for (int v = 0; v < 8; ++v) {
    rs0[v] = rowScale[v + mofs];       of0[v] = outOfs[v + mofs];
    rs1[v] = rowScale[16 + v + mofs];  of1[v] = outOfs[16 + v + mofs];
  }

  const size_t ebase = (size_t)bs * (LL * DD);

  // ---- main loop over D chunks ------------------------------------------
  for (int c = 0; c < NCH; ++c) {
    const int d0 = c * DC;

    // stage emb chunk -> bf16 transposed LDS tile (coalesced global reads,
    // conflict-free LDS column writes)
    for (int i = t; i < LL * DC; i += 256) {
      int l = i >> 7, dc = i & 127;
      embT[dc * RST + l] = f2bf(emb_g[ebase + (size_t)l * DD + (d0 + dc)]);
    }
    __syncthreads();

    // prefetch next chunk into caches (global_prefetch_b8)
    if (c + 1 < NCH) {
      const float* nx = emb_g + ebase + (size_t)(c + 1) * DC;
#pragma unroll
      for (int j = 0; j < 2; ++j) {
        int line = t * 2 + j;                 // 512 x 128B lines cover chunk
        int l = line >> 2, off = (line & 3) * 32;
        __builtin_prefetch(nx + (size_t)l * DD + off, 0, 3);
      }
    }

    // GEMM: wave w owns N-tile w; both M-tiles share each B fragment
    const unsigned int* brow =
        (const unsigned int*)(embT + (w * 16 + (lane & 15)) * RST);
    v8f acc0 = {0.f,0.f,0.f,0.f,0.f,0.f,0.f,0.f};
    v8f acc1 = {0.f,0.f,0.f,0.f,0.f,0.f,0.f,0.f};
#pragma unroll
    for (int kk = 0; kk < 4; ++kk) {
      v8ui bu;
      int kb = (kk * 32 + g * 8) >> 1;        // dword index into padded row
#pragma unroll
      for (int j = 0; j < 8; ++j) bu[j] = brow[kb + j + ((j >= 4) ? 4 : 0)];
      v16bf bfb = __builtin_bit_cast(v16bf, bu);
      acc0 = __builtin_amdgcn_wmma_f32_16x16x32_bf16(
          false, afrag[0][kk], false, bfb, (short)0, acc0, false, false);
      acc1 = __builtin_amdgcn_wmma_f32_16x16x32_bf16(
          false, afrag[1][kk], false, bfb, (short)0, acc1, false, false);
    }

    // scale + scatter C rows (all address math precomputed)
    const int n = d0 + w * 16 + (lane & 15);
#pragma unroll
    for (int v = 0; v < 8; ++v) {
      out[of0[v] + n] = acc0[v] * rs0[v];                 // rows 0..15
      if (v != 7 || lane < 16)                            // skip pad row 31
        out[of1[v] + n] = acc1[v] * rs1[v];               // rows 16..30
    }
    __syncthreads();
  }
}

extern "C" void kernel_launch(void* const* d_in, const int* in_sizes, int n_in,
                              void* d_out, int out_size, void* d_ws, size_t ws_size,
                              hipStream_t stream) {
  (void)in_sizes; (void)n_in; (void)d_ws; (void)ws_size; (void)out_size;
  const int*   sids = (const int*)d_in[0];
  const float* mask = (const float*)d_in[1];
  const float* emb  = (const float*)d_in[2];
  const int*   pred = (const int*)d_in[3];
  const int*   a0   = (const int*)d_in[4];
  const int*   a1   = (const int*)d_in[5];
  float* out = (float*)d_out;
  hipLaunchKernelGGL(srl_fused, dim3(NBS), dim3(256), 0, stream,
                     sids, mask, emb, pred, a0, a1, out);
}
